// SlideWindowConvolution_60911226192298
// MI455X (gfx1250) — matrix operand order repaired and checked by reading, not compiled, not run on hardware
//
#include <hip/hip_runtime.h>

#ifndef __has_builtin
#define __has_builtin(x) 0
#endif

#if __has_builtin(__builtin_amdgcn_global_load_async_to_lds_b128) && \
    __has_builtin(__builtin_amdgcn_s_wait_asynccnt)
#define USE_ASYNC_LDS 1
#else
#define USE_ASYNC_LDS 0
#endif

typedef __attribute__((ext_vector_type(16))) __bf16 v16bf;
typedef __attribute__((ext_vector_type(8)))  __bf16 v8bf;
typedef __attribute__((ext_vector_type(8)))  float  v8f;
typedef __attribute__((ext_vector_type(4)))  int    v4i;   // async-LDS builtin pointee type

#define B_    8
#define C_    64
#define O_    64
#define H_    112
#define W_    112
#define HW_   (H_*W_)          // 12544
#define KT_   576              // 9 taps * 64 channels, tap-major: k = tap*64 + c
#define NPOS_ (B_*HW_)         // 100352
#define BLK_N 128
#define KSTEP 64               // one tap, all 64 channels
#define LROW  72               // padded LDS row stride (bf16); 144B rows stay 16B aligned
#define CONV_SIZE (B_*O_*HW_)  // 6422528

// ---- prep: weights fp32 (O,C,3,3) -> bf16 [oc][tap*64+c] (tap-major K) + per-(o,c) zero masks
__global__ void prep_kernel(const float* __restrict__ w, __bf16* __restrict__ wbf,
                            unsigned* __restrict__ wmask) {
    int idx = blockIdx.x * blockDim.x + threadIdx.x;      // 0..36863
    if (idx < O_ * KT_) {
        int oc  = idx / KT_;
        int rem = idx - oc * KT_;
        int tap = rem >> 6;
        int c   = rem & 63;
        wbf[idx] = (__bf16)w[(oc * C_ + c) * 9 + tap];
    }
    if (idx < O_ * C_) {
        const float* wp = w + idx * 9;
        unsigned m = 0;
        #pragma unroll
        for (int t = 0; t < 9; ++t) m |= (wp[t] == 0.0f ? 1u : 0u) << t;
        wmask[idx] = m;
    }
}

// ---- conv as implicit GEMM with bf16 WMMA (f32 accumulate), tap-major K streaming ----------
__global__ __launch_bounds__(256)
void conv_wmma_kernel(const float* __restrict__ x, const __bf16* __restrict__ wbf,
                      const float* __restrict__ bias, float* __restrict__ out) {
    __shared__ __bf16 w_lds[O_ * LROW];      // 64 x 64 bf16 tile (padded)   9216 B
    __shared__ __bf16 p_lds[BLK_N * LROW];   // 128 x 64 bf16 tile (padded) 18432 B

    const int tid   = threadIdx.x;
    const int lane  = tid & 31;
    const int wave  = tid >> 5;
    const int n0    = blockIdx.x * BLK_N;

    const int mtile = wave & 3;          // 16 output channels per wave
    const int ntg   = (wave >> 2) * 4;   // four 16-col N tiles per wave
    const int col   = lane & 15;
    const int khalf = lane >> 4;

    // ---- gather geometry: one spatial position per thread, hoisted out of the K loop ----
    const int nlocal = tid & 127;
    const int chunk  = tid >> 7;         // 0/1: which 32-channel half this thread fills
    const int n  = n0 + nlocal;
    const int bb = n / HW_;
    const int rr = n - bb * HW_;
    const int hh = rr / W_;
    const int ww = rr - hh * W_;
    const float* xrow = x + ((size_t)bb * C_ * H_ + hh) * W_ + ww;   // &x[b][0][h][w]

    v8f acc[4];
    #pragma unroll
    for (int nt = 0; nt < 4; ++nt)
        #pragma unroll
        for (int i = 0; i < 8; ++i) acc[nt][i] = 0.0f;

    for (int tap = 0; tap < 9; ++tap) {
        __syncthreads();

        // ---- stage 64x64 bf16 weight tile: 512 x b128; async DMA to LDS when available ----
        #pragma unroll
        for (int q0 = 0; q0 < 2; ++q0) {
            int q  = tid + q0 * 256;
            int oc = q >> 3;
            int kg = (q & 7) * 8;
#if USE_ASYNC_LDS
            __builtin_amdgcn_global_load_async_to_lds_b128(
                (v4i*)(wbf + oc * KT_ + tap * KSTEP + kg),
                (v4i*)&w_lds[oc * LROW + kg], 0, 0);
#else
            *(uint4*)&w_lds[oc * LROW + kg] =
                *(const uint4*)&wbf[oc * KT_ + tap * KSTEP + kg];
#endif
        }

        // ---- stage 128x64 bf16 patch tile: uniform (dy,dx) per tap, stride-HW channel walk --
        {
            const int dyq = tap / 3;
            const int dy  = dyq - 1;
            const int dx  = tap - dyq * 3 - 1;
            const bool ok = ((unsigned)(hh + dy) < (unsigned)H_) &&
                            ((unsigned)(ww + dx) < (unsigned)W_);
            const float* src = xrow + dy * W_ + dx + chunk * 32 * HW_;
            v8bf t0, t1, t2, t3;
            if (ok) {
                #pragma unroll
                for (int i = 0; i < 8; ++i) {
                    t0[i] = (__bf16)src[(i +  0) * HW_];
                    t1[i] = (__bf16)src[(i +  8) * HW_];
                    t2[i] = (__bf16)src[(i + 16) * HW_];
                    t3[i] = (__bf16)src[(i + 24) * HW_];
                }
            } else {
                #pragma unroll
                for (int i = 0; i < 8; ++i) {
                    t0[i] = (__bf16)0.0f; t1[i] = (__bf16)0.0f;
                    t2[i] = (__bf16)0.0f; t3[i] = (__bf16)0.0f;
                }
            }
            __bf16* dst = &p_lds[nlocal * LROW + chunk * 32];
            *(v8bf*)(dst +  0) = t0;
            *(v8bf*)(dst +  8) = t1;
            *(v8bf*)(dst + 16) = t2;
            *(v8bf*)(dst + 24) = t3;
        }

#if USE_ASYNC_LDS
        __builtin_amdgcn_s_wait_asynccnt(0);
#endif
        __syncthreads();

        // ---- 2 K-chunks of 32 per tap: 8 WMMAs per wave per staged tile ----
        #pragma unroll
        for (int kc = 0; kc < 2; ++kc) {
            v16bf a;   // A 16x32: lanes 0-15 M=0..15; K = khalf*16 + [0..16)
            const int ab = (mtile * 16 + col) * LROW + kc * 32 + khalf * 16;
            ((v8bf*)&a)[0] = *(const v8bf*)&w_lds[ab];
            ((v8bf*)&a)[1] = *(const v8bf*)&w_lds[ab + 8];

            #pragma unroll
            for (int nt = 0; nt < 4; ++nt) {
                const int bi = ((ntg + nt) * 16 + col) * LROW + kc * 32 + khalf * 16;
                v16bf bfr;  // B 32x16: N = col; K = khalf*16 + [0..16)
                ((v8bf*)&bfr)[0] = *(const v8bf*)&p_lds[bi];
                ((v8bf*)&bfr)[1] = *(const v8bf*)&p_lds[bi + 8];

                acc[nt] = __builtin_amdgcn_wmma_f32_16x16x32_bf16(
                    false, a, false, bfr, (short)0, acc[nt], false, false);
            }
        }
    }

    // ---- epilogue: C/D layout — elem r: M = r + 8*khalf, N = col ----
    #pragma unroll
    for (int nt = 0; nt < 4; ++nt) {
        int nn = n0 + (ntg + nt) * 16 + col;
        int b2 = nn / HW_;
        int r2 = nn - b2 * HW_;
        int h2 = r2 / W_;
        int w2 = r2 - h2 * W_;
        #pragma unroll
        for (int e = 0; e < 8; ++e) {
            int oc = mtile * 16 + e + khalf * 8;
            out[((b2 * O_ + oc) * H_ + h2) * W_ + w2] = acc[nt][e] + bias[oc];
        }
    }
}

// ---- stage 1: per-(c,b) plane, histogram 9-bit patch-zero masks over positions --------------
__global__ __launch_bounds__(256)
void zero_mask_hist_kernel(const float* __restrict__ x, unsigned* __restrict__ hist_mask) {
    __shared__ unsigned lh[512];
    const int tid = threadIdx.x;
    for (int i = tid; i < 512; i += 256) lh[i] = 0u;
    __syncthreads();

    const int cb = blockIdx.y;           // 0..511 -> (c, b)
    const int c  = cb & 63;
    const int b  = cb >> 6;
    const int pos = blockIdx.x * 256 + tid;   // 49*256 == 12544 exactly
    const int h = pos / W_;
    const int w = pos - h * W_;
    const float* plane = x + (size_t)(b * C_ + c) * HW_;

    unsigned m = 0;
    int bit = 0;
    #pragma unroll
    for (int dy = -1; dy <= 1; ++dy)
        #pragma unroll
        for (int dx = -1; dx <= 1; ++dx) {
            int y = h + dy, xx = w + dx;
            float v = 0.0f;
            if ((unsigned)y < (unsigned)H_ && (unsigned)xx < (unsigned)W_)
                v = plane[y * W_ + xx];
            m |= (v == 0.0f ? 1u : 0u) << bit;
            ++bit;
        }
    atomicAdd(&lh[m], 1u);
    __syncthreads();
    for (int i = tid; i < 512; i += 256)
        if (lh[i]) atomicAdd(&hist_mask[c * 512 + i], lh[i]);
}

// ---- stage 2: expand mask histogram over out-channels analytically --------------------------
__global__ void hist_final_kernel(const unsigned* __restrict__ hist_mask,
                                  const unsigned* __restrict__ wmask,
                                  float* __restrict__ hist) {
    int c = threadIdx.x;
    if (c >= C_) return;
    float acc[10];
    #pragma unroll
    for (int z = 0; z < 10; ++z) acc[z] = 0.0f;
    for (int m = 0; m < 512; ++m) {
        unsigned cnt = hist_mask[c * 512 + m];
        if (!cnt) continue;
        float fc = (float)cnt;
        for (int o = 0; o < O_; ++o) {
            int z = __popc((unsigned)m | wmask[o * C_ + c]);   // zeros in product window
            acc[z] += fc;
        }
    }
    for (int z = 0; z < 10; ++z) hist[c * 10 + z] = acc[z];
}

extern "C" void kernel_launch(void* const* d_in, const int* in_sizes, int n_in,
                              void* d_out, int out_size, void* d_ws, size_t ws_size,
                              hipStream_t stream) {
    (void)in_sizes; (void)n_in; (void)out_size; (void)ws_size;
    const float* x    = (const float*)d_in[0];
    const float* wgt  = (const float*)d_in[1];
    const float* bias = (const float*)d_in[2];
    float* conv = (float*)d_out;
    float* hist = conv + CONV_SIZE;

    char* ws = (char*)d_ws;
    __bf16*   wbf   = (__bf16*)ws;                         // 73728 B
    unsigned* wmask = (unsigned*)(ws + 73728);             // 16384 B
    unsigned* hmask = (unsigned*)(ws + 73728 + 16384);     // 131072 B

    (void)hipMemsetAsync(hmask, 0, 64 * 512 * sizeof(unsigned), stream);
    prep_kernel<<<144, 256, 0, stream>>>(wgt, wbf, wmask);
    conv_wmma_kernel<<<NPOS_ / BLK_N, 256, 0, stream>>>(x, wbf, bias, conv);
    zero_mask_hist_kernel<<<dim3(49, 512), 256, 0, stream>>>(x, hmask);
    hist_final_kernel<<<1, 64, 0, stream>>>(hmask, wmask, hist);
}
